// PointerNetwork_28613072126346
// MI455X (gfx1250) — matrix operand order repaired and checked by reading, not compiled, not run on hardware
//
#include <hip/hip_runtime.h>

typedef __attribute__((ext_vector_type(16))) __bf16 v16bf;
typedef __attribute__((ext_vector_type(8)))  __bf16 v8bf;
typedef __attribute__((ext_vector_type(8)))  float  v8f;

#define B_   512
#define L_   256
#define H_   512
#define H4_  2048
#define LP1  257   // L_ + 1 (slot 0 = zero initial hidden state)

// ---------- small helpers ----------
static __device__ __forceinline__ unsigned short f2bf(float f) {
  unsigned u = __builtin_bit_cast(unsigned, f);
  unsigned r = u + 0x7FFFu + ((u >> 16) & 1u);   // round-to-nearest-even
  return (unsigned short)(r >> 16);
}
static __device__ __forceinline__ float bf2f(unsigned short s) {
  unsigned u = ((unsigned)s) << 16;
  return __builtin_bit_cast(float, u);
}
static __device__ __forceinline__ float sigmoidf(float x) {
  return 1.0f / (1.0f + __expf(-x));
}
static __device__ __forceinline__ v8f wmma_bf16(v16bf a, v16bf b, v8f c) {
  return __builtin_amdgcn_wmma_f32_16x16x32_bf16(false, a, false, b, (short)0, c,
                                                 false, false);
}
// A-matrix 16x32 bf16 fragment per CDNA5 layout:
// lanes 0-15: M=lane, K = k+{0..7} (VGPR0-3) and k+{16..23} (VGPR4-7)
// lanes 16-31: M=lane-16, K = k+{8..15} and k+{24..31}
static __device__ __forceinline__ v16bf load_a_frag(const unsigned short* row,
                                                    int k, int lane) {
  int kh = (lane >> 4) << 3;
  union { v16bf v; v8bf h[2]; } u;
  u.h[0] = *(const v8bf*)(row + k + kh);
  u.h[1] = *(const v8bf*)(row + k + kh + 16);
  return u.v;
}
// B-matrix 32x16 bf16 fragment: lane&15 = N column; lanes 0-15 hold K=k+{0..7}
// and k+{16..23}, lanes 16-31 hold K=k+{8..15} and k+{24..31} (same as A)
// from the row-major (N,K) weight matrix.
static __device__ __forceinline__ v16bf load_b_frag(const unsigned short* brow,
                                                    int k, int lane) {
  int kh = (lane >> 4) << 3;
  union { v16bf v; v8bf h[2]; } u;
  u.h[0] = *(const v8bf*)(brow + k + kh);
  u.h[1] = *(const v8bf*)(brow + k + kh + 16);
  return u.v;
}

// ---------- LSTM cell step ----------
// gates = x_m*r[n] + s[n] + h_prev @ Whh^T + bih[n] + bhh[n]
// (embedding GEMM folded to rank-1: r = Wih@embW, s = Wih@embB, F == 1)
// Each wave: 16 batch rows x 16 gate-j columns, computing all 4 gates
// (columns j, j+H, j+2H, j+3H) so the activation epilogue is wave-local.
__global__ __launch_bounds__(256) void lstm_cell_kernel(
    const float* __restrict__ xvec, int xstride,                  // scalar input per row
    const float* __restrict__ rvec, const float* __restrict__ svec, // (4H)
    const unsigned short* __restrict__ hprev, long hstride,       // bf16 (B x H)
    const unsigned short* __restrict__ Whh,                       // bf16 (4H x H)
    const float* __restrict__ bih, const float* __restrict__ bhh, // (4H)
    float* __restrict__ cbuf,                                     // f32 (B x H) in/out
    unsigned short* __restrict__ hout, long hostride)             // bf16 (B x H) out
{
  int wid  = (blockIdx.x * blockDim.x + threadIdx.x) >> 5;
  int lane = threadIdx.x & 31;
  int mt = wid & 31, jt = wid >> 5;           // 32 x 32 wave tiles
  int m0 = mt << 4, j0 = jt << 4;

  int mA   = m0 + (lane & 15);
  int ncol = j0 + (lane & 15);

  const unsigned short* whh_rows[4];
#pragma unroll
  for (int g = 0; g < 4; g++)
    whh_rows[g] = Whh + (size_t)(g * H_ + ncol) * H_;

  v8f acc[4] = {};

  // h_prev @ Whh^T on the matrix cores
  const unsigned short* arow = hprev + (size_t)mA * hstride;
  for (int k = 0; k < H_; k += 32) {
    v16bf af = load_a_frag(arow, k, lane);
#pragma unroll
    for (int g = 0; g < 4; g++)
      acc[g] = wmma_bf16(af, load_b_frag(whh_rows[g], k, lane), acc[g]);
  }

  // rank-1 input path + biases
  float rg[4], base[4];
#pragma unroll
  for (int g = 0; g < 4; g++) {
    int n = g * H_ + ncol;
    rg[g]   = rvec[n];
    base[g] = svec[n] + bih[n] + bhh[n];
  }
  float xm[8];
#pragma unroll
  for (int r = 0; r < 8; r++)
    xm[r] = xvec[(size_t)(m0 + r + ((lane >> 4) << 3)) * xstride];

#pragma unroll
  for (int r = 0; r < 8; r++) {
    int m = m0 + r + ((lane >> 4) << 3);     // C/D layout: VGPR r -> row r / r+8
    float iv = sigmoidf(acc[0][r] + xm[r] * rg[0] + base[0]);
    float fv = sigmoidf(acc[1][r] + xm[r] * rg[1] + base[1]);
    float gv = tanhf  (acc[2][r] + xm[r] * rg[2] + base[2]);
    float ov = sigmoidf(acc[3][r] + xm[r] * rg[3] + base[3]);
    size_t coff = (size_t)m * H_ + ncol;
    float cn = fv * cbuf[coff] + iv * gv;
    cbuf[coff] = cn;
    hout[(size_t)m * hostride + ncol] = f2bf(ov * tanhf(cn));
  }
}

// ---------- generic C = A @ Bw^T + bias (A bf16 strided, Bw (N,K) bf16) ----------
template <bool OUT_BF16>
__global__ __launch_bounds__(256) void gemm_bias_kernel(
    const unsigned short* __restrict__ A, long lda,
    const unsigned short* __restrict__ Bw, int ldb,
    const float* __restrict__ bias,
    void* __restrict__ Cout, long ldc,
    int Mtiles, int Ntiles, int K)
{
  int wid  = (blockIdx.x * blockDim.x + threadIdx.x) >> 5;
  int lane = threadIdx.x & 31;
  int mt = wid / Ntiles, nt = wid - mt * Ntiles;
  if (mt >= Mtiles) return;
  int m0 = mt << 4, n0 = nt << 4;
  const unsigned short* arow = A  + (size_t)(m0 + (lane & 15)) * lda;
  const unsigned short* brow = Bw + (size_t)(n0 + (lane & 15)) * ldb;
  v8f acc = {};
  for (int k = 0; k < K; k += 32)
    acc = wmma_bf16(load_a_frag(arow, k, lane), load_b_frag(brow, k, lane), acc);
  float bv = bias[n0 + (lane & 15)];
#pragma unroll
  for (int r = 0; r < 8; r++) {
    float v = acc[r] + bv;
    size_t off = (size_t)(m0 + r + ((lane >> 4) << 3)) * ldc + (n0 + (lane & 15));
    if (OUT_BF16) ((unsigned short*)Cout)[off] = f2bf(v);
    else          ((float*)Cout)[off] = v;
  }
}

// ---------- attention + log-softmax + argmax pointer (one block per batch) ----------
__global__ __launch_bounds__(256) void attention_kernel(
    const unsigned short* __restrict__ encp,   // bf16 (B, LP1, H), slot l+1 = enc step l
    const float* __restrict__ q,               // f32 (B, H)
    const float* __restrict__ vW, const float* __restrict__ vb,
    const float* __restrict__ input_seq,       // (B, L)
    float* __restrict__ attn_out, float* __restrict__ ptr_out,
    float* __restrict__ nxt, int t)
{
  __shared__ float qs[H_], vs[H_];
  __shared__ float red[L_];
  __shared__ int   ridx[L_];
  int b = blockIdx.x, l = threadIdx.x;
  qs[l]       = q[(size_t)b * H_ + l];
  qs[l + 256] = q[(size_t)b * H_ + l + 256];
  vs[l]       = vW[l];
  vs[l + 256] = vW[l + 256];
  __syncthreads();

  const unsigned short* row = encp + ((size_t)b * LP1 + 1 + l) * H_;
  float u = vb[0];
  for (int h = 0; h < H_; h += 8) {
    uint4 raw = *(const uint4*)(row + h);
    unsigned p;
    p = raw.x;
    u += tanhf(bf2f((unsigned short)(p & 0xffffu)) + qs[h + 0]) * vs[h + 0];
    u += tanhf(bf2f((unsigned short)(p >> 16))     + qs[h + 1]) * vs[h + 1];
    p = raw.y;
    u += tanhf(bf2f((unsigned short)(p & 0xffffu)) + qs[h + 2]) * vs[h + 2];
    u += tanhf(bf2f((unsigned short)(p >> 16))     + qs[h + 3]) * vs[h + 3];
    p = raw.z;
    u += tanhf(bf2f((unsigned short)(p & 0xffffu)) + qs[h + 4]) * vs[h + 4];
    u += tanhf(bf2f((unsigned short)(p >> 16))     + qs[h + 5]) * vs[h + 5];
    p = raw.w;
    u += tanhf(bf2f((unsigned short)(p & 0xffffu)) + qs[h + 6]) * vs[h + 6];
    u += tanhf(bf2f((unsigned short)(p >> 16))     + qs[h + 7]) * vs[h + 7];
  }

  // argmax (first occurrence wins ties) + max
  red[l] = u; ridx[l] = l; __syncthreads();
  for (int s = 128; s > 0; s >>= 1) {
    if (l < s) {
      float a = red[l], c2 = red[l + s];
      int ia = ridx[l], ic = ridx[l + s];
      if (c2 > a || (c2 == a && ic < ia)) { red[l] = c2; ridx[l] = ic; }
    }
    __syncthreads();
  }
  float umax = red[0]; int amax = ridx[0];
  __syncthreads();
  red[l] = __expf(u - umax); __syncthreads();
  for (int s = 128; s > 0; s >>= 1) {
    if (l < s) red[l] += red[l + s];
    __syncthreads();
  }
  float lse = __logf(red[0]);

  attn_out[((size_t)b * L_ + t) * L_ + l] = u - umax - lse;
  if (l == 0) {
    ptr_out[(size_t)b * L_ + t] = (float)amax;
    nxt[b] = input_seq[(size_t)b * L_ + amax];
  }
}

// ---------- r = W @ v0, s = W @ v1 (one wave per row, f32) ----------
__global__ __launch_bounds__(256) void matvec2_kernel(
    const float* __restrict__ W,                                  // (N x K) f32
    const float* __restrict__ v0, const float* __restrict__ v1,   // (K)
    float* __restrict__ r0, float* __restrict__ r1, int N, int K)
{
  int wid  = (blockIdx.x * blockDim.x + threadIdx.x) >> 5;
  int lane = threadIdx.x & 31;
  if (wid >= N) return;
  const float* row = W + (size_t)wid * K;
  float s0 = 0.f, s1 = 0.f;
  for (int k = lane; k < K; k += 32) {
    float wv = row[k];
    s0 += wv * v0[k];
    s1 += wv * v1[k];
  }
#pragma unroll
  for (int m = 16; m; m >>= 1) {
    s0 += __shfl_xor(s0, m, 32);
    s1 += __shfl_xor(s1, m, 32);
  }
  if (lane == 0) { r0[wid] = s0; r1[wid] = s1; }
}

// ---------- utility kernels ----------
__global__ void cvt_f32_bf16(const float* __restrict__ s, unsigned short* __restrict__ d, int n) {
  for (int i = blockIdx.x * blockDim.x + threadIdx.x; i < n; i += gridDim.x * blockDim.x)
    d[i] = f2bf(s[i]);
}
__global__ void fill_f32(float* __restrict__ p, float v, int n) {
  int i = blockIdx.x * blockDim.x + threadIdx.x;
  if (i < n) p[i] = v;
}
__global__ void zero_h0(unsigned short* __restrict__ henc) {   // zero slot 0 per batch
  int i = blockIdx.x * blockDim.x + threadIdx.x;               // 0 .. B*H-1
  int b = i >> 9, h = i & 511;
  henc[(size_t)b * LP1 * H_ + h] = 0;
}

extern "C" void kernel_launch(void* const* d_in, const int* in_sizes, int n_in,
                              void* d_out, int out_size, void* d_ws, size_t ws_size,
                              hipStream_t stream)
{
  const float* in_seq  = (const float*)d_in[0];
  const float* eembW   = (const float*)d_in[1];
  const float* eembB   = (const float*)d_in[2];
  const float* eWih    = (const float*)d_in[3];
  const float* eWhh    = (const float*)d_in[4];
  const float* ebih    = (const float*)d_in[5];
  const float* ebhh    = (const float*)d_in[6];
  const float* dembW   = (const float*)d_in[7];
  const float* dembB   = (const float*)d_in[8];
  const float* dWih    = (const float*)d_in[9];
  const float* dWhh    = (const float*)d_in[10];
  const float* dbih    = (const float*)d_in[11];
  const float* dbhh    = (const float*)d_in[12];
  const float* W1      = (const float*)d_in[13];
  const float* W1b     = (const float*)d_in[14];
  const float* W2      = (const float*)d_in[15];
  const float* W2b     = (const float*)d_in[16];
  const float* vW      = (const float*)d_in[17];
  const float* vb      = (const float*)d_in[18];

  char* w = (char*)d_ws;
  size_t off = 0;
  auto alloc = [&](size_t bytes) -> char* {
    char* p = w + off;
    off += (bytes + 255) & ~(size_t)255;
    return p;
  };
  unsigned short* eWhh_bf = (unsigned short*)alloc((size_t)H4_ * H_ * 2);
  unsigned short* dWhh_bf = (unsigned short*)alloc((size_t)H4_ * H_ * 2);
  unsigned short* W1_bf   = (unsigned short*)alloc((size_t)H_ * H_ * 2);
  unsigned short* W2_bf   = (unsigned short*)alloc((size_t)H_ * H_ * 2);
  unsigned short* henc    = (unsigned short*)alloc((size_t)B_ * LP1 * H_ * 2); // bf16 h history
  unsigned short* encp    = (unsigned short*)alloc((size_t)B_ * LP1 * H_ * 2); // bf16 W1(enc_out), L2-resident
  float*          cbuf    = (float*)alloc((size_t)B_ * H_ * 4);
  unsigned short* hdec0   = (unsigned short*)alloc((size_t)B_ * H_ * 2);
  unsigned short* hdec1   = (unsigned short*)alloc((size_t)B_ * H_ * 2);
  float*          qbuf    = (float*)alloc((size_t)B_ * H_ * 4);
  float*          nxt     = (float*)alloc((size_t)B_ * 4);
  float*          renc    = (float*)alloc((size_t)H4_ * 4);   // eWih @ eembW
  float*          senc    = (float*)alloc((size_t)H4_ * 4);   // eWih @ eembB
  float*          rdec    = (float*)alloc((size_t)H4_ * 4);   // dWih @ dembW
  float*          sdec    = (float*)alloc((size_t)H4_ * 4);   // dWih @ dembB

  // recurrent weights -> bf16 (all stay L2-resident: <= 2 MB each)
  cvt_f32_bf16<<<2048, 256, 0, stream>>>(eWhh, eWhh_bf, H4_ * H_);
  cvt_f32_bf16<<<2048, 256, 0, stream>>>(dWhh, dWhh_bf, H4_ * H_);
  cvt_f32_bf16<<<1024, 256, 0, stream>>>(W1, W1_bf, H_ * H_);
  cvt_f32_bf16<<<1024, 256, 0, stream>>>(W2, W2_bf, H_ * H_);

  // rank-1 input path: fold the F==1 embedding + Wih GEMM into two vectors
  matvec2_kernel<<<H4_ / 8, 256, 0, stream>>>(eWih, eembW, eembB, renc, senc, H4_, H_);
  matvec2_kernel<<<H4_ / 8, 256, 0, stream>>>(dWih, dembW, dembB, rdec, sdec, H4_, H_);

  fill_f32<<<(B_ * H_) / 256, 256, 0, stream>>>(cbuf, 0.0f, B_ * H_);
  zero_h0<<<(B_ * H_) / 256, 256, 0, stream>>>(henc);
  fill_f32<<<2, 256, 0, stream>>>(nxt, -1.0f, B_);          // SOS

  const long hencStride = (long)LP1 * H_;

  // ---- encoder: 256 sequential cell steps (stream-ordered) ----
  for (int t = 0; t < L_; t++) {
    lstm_cell_kernel<<<128, 256, 0, stream>>>(
        in_seq + t, L_,                 // x[b] = input_seq[b, t, 0]
        renc, senc,
        henc + (size_t)t * H_, hencStride,
        eWhh_bf, ebih, ebhh,
        cbuf,
        henc + (size_t)(t + 1) * H_, hencStride);
  }

  // ---- enc_proj = W1(enc_out) + b, stored bf16 (fits in 192 MB L2) ----
  {
    int Mtiles = (B_ * LP1) / 16;       // 8224 (slot 0 rows are zeros, harmless)
    int Ntiles = H_ / 16;               // 32
    int blocks = (Mtiles * Ntiles) / 8; // 8 waves per block
    gemm_bias_kernel<true><<<blocks, 256, 0, stream>>>(
        henc, H_, W1_bf, H_, W1b, encp, H_, Mtiles, Ntiles, H_);
  }

  // ---- decoder: cell -> q -> attention/pointer, 256 steps ----
  for (int t = 0; t < L_; t++) {
    const unsigned short* hprev = (t == 0) ? (henc + (size_t)L_ * H_)
                                           : ((t & 1) ? hdec0 : hdec1);
    long hs = (t == 0) ? hencStride : (long)H_;
    unsigned short* hcur = (t & 1) ? hdec1 : hdec0;

    lstm_cell_kernel<<<128, 256, 0, stream>>>(
        nxt, 1, rdec, sdec, hprev, hs,
        dWhh_bf, dbih, dbhh,
        cbuf, hcur, (long)H_);

    gemm_bias_kernel<false><<<128, 256, 0, stream>>>(
        hcur, H_, W2_bf, H_, W2b, qbuf, H_, B_ / 16, H_ / 16, H_);

    attention_kernel<<<B_, L_, 0, stream>>>(
        encp, qbuf, vW, vb, in_seq,
        (float*)d_out,
        (float*)d_out + (size_t)B_ * L_ * L_,
        nxt, t);
  }
}
